// SamplingDecoder_26259430048173
// MI455X (gfx1250) — matrix operand order repaired and checked by reading, not compiled, not run on hardware
//
#include <hip/hip_runtime.h>
#include <hip/hip_bf16.h>

#define B_   64
#define E_   1024
#define H_   1024
#define H3_  3072
#define V_   32000
#define T_   32
#define SOS_TOK 1
#define EOS_TOK 2
#define NV_BLK 256            // vocab columns per logits block
#define NBLK_V (V_ / NV_BLK)  // 125

typedef __attribute__((ext_vector_type(16))) __bf16 v16bf;
typedef __attribute__((ext_vector_type(8)))  __bf16 v8bf;
typedef __attribute__((ext_vector_type(8)))  float  v8f;

__device__ __forceinline__ v8f zero8() {
  v8f z;
#pragma unroll
  for (int i = 0; i < 8; ++i) z[i] = 0.0f;
  return z;
}

// A-matrix 16x32 bf16 tile (rows of activation matrix X, row-major, leading dim ld).
// Layout: lanes 0-15 rows M=0..15; element s = K[hi*16 + s] (contiguous, matches B).
__device__ __forceinline__ v16bf load_a16x32(const __bf16* __restrict__ X, int ld,
                                             int m0, int k0, int lane) {
  int m  = m0 + (lane & 15);
  int hi = lane >> 4;
  const __bf16* p = X + (size_t)m * ld + k0 + hi * 16;
  return *(const v16bf*)(p);
}

// B-matrix 32x16 bf16 tile for X @ W^T: B[k][n] = W[n][k], W row-major [N,K] (ld=K).
// Lane's 16 K-values for its column n are 16 contiguous bf16 in W's row n.
__device__ __forceinline__ v16bf load_b32x16(const __bf16* __restrict__ W, int ld,
                                             int n0, int k0, int lane) {
  int n  = n0 + (lane & 15);
  int hi = lane >> 4;
  return *(const v16bf*)(W + (size_t)n * ld + k0 + hi * 16);
}

__device__ __forceinline__ v8f wmma_bf16(v16bf a, v16bf b, v8f c) {
  return __builtin_amdgcn_wmma_f32_16x16x32_bf16(false, a, false, b, (short)0, c,
                                                 false, false);
}

// Async DMA: 16 bytes per lane, global memory -> LDS, tracked by ASYNCcnt.
// VDST = per-lane LDS byte offset, VADDR = per-lane 64-bit global address.
__device__ __forceinline__ void async_b128_to_lds(unsigned lds_off, const void* gsrc) {
  asm volatile("global_load_async_to_lds_b128 %0, %1, off"
               :: "v"(lds_off), "v"(gsrc)
               : "memory");
}
__device__ __forceinline__ void wait_asynccnt0() {
  asm volatile("s_wait_asynccnt 0x0" ::: "memory");
}

// One full-K GEMM pass: acc{0,1} += A[m0:m0+16, :] * W[nbase:nbase+32, :]^T.
// unroll 1: keep the in-flight load window small; latency is hidden by the
// 8 resident waves, not by register-hungry prefetch windows (avoids spills).
__device__ __forceinline__ void gemm_pass(const __bf16* __restrict__ A, int lda,
                                          const __bf16* __restrict__ W, int ldw,
                                          int nbase, int m0, int lane,
                                          v8f& acc0, v8f& acc1) {
#pragma unroll 1
  for (int k0 = 0; k0 < H_; k0 += 32) {
    v16bf a  = load_a16x32(A, lda, m0, k0, lane);
    v16bf b0 = load_b32x16(W, ldw, nbase,      k0, lane);
    v16bf b1 = load_b32x16(W, ldw, nbase + 16, k0, lane);
    acc0 = wmma_bf16(a, b0, acc0);
    acc1 = wmma_bf16(a, b1, acc1);
  }
}

// ---------------------------------------------------------------------------
// fp32 -> bf16 weight conversion (one-time per launch; weights then L2-resident)
// ---------------------------------------------------------------------------
__global__ void cvt_f32_bf16_kernel(const float* __restrict__ src,
                                    __bf16* __restrict__ dst, long n4) {
  long i = (long)blockIdx.x * blockDim.x + threadIdx.x;
  if (i < n4) {
    float4 v = ((const float4*)src)[i];
    __bf16* d = dst + i * 4;
    d[0] = (__bf16)v.x; d[1] = (__bf16)v.y;
    d[2] = (__bf16)v.z; d[3] = (__bf16)v.w;
  }
}

// h0 = concat(zs, cs); x0 = emb[SOS]
__global__ void init_state_kernel(const float* __restrict__ zs,
                                  const float* __restrict__ cs,
                                  const float* __restrict__ emb,
                                  float* __restrict__ hf,
                                  __bf16* __restrict__ h0,
                                  __bf16* __restrict__ x) {
  int i = blockIdx.x * blockDim.x + threadIdx.x;
  if (i >= B_ * H_) return;
  int b = i >> 10, j = i & (H_ - 1);
  float hv = (j < 512) ? zs[b * 512 + j] : cs[b * 512 + (j - 512)];
  hf[i] = hv;
  h0[i] = (__bf16)hv;
  x[i]  = (__bf16)emb[(size_t)SOS_TOK * E_ + j];
}

// ---------------------------------------------------------------------------
// One GRU step for a 64-column slice of the hidden state.
// Block = 256 threads (8 waves) = exactly 2 waves/SIMD32 -> pin waves_per_eu=2
// so the allocator can use the full per-wave VGPR budget instead of spilling.
// Wave w: M-tile mt = w>>1 (batch rows), two 16-wide N-tiles at nt0 = (w&1)*2.
// Gates run as sequential full-K passes so only one acc pair is live at a time;
// r,z chain the ih and hh GEMMs into the same accumulator.
// ---------------------------------------------------------------------------
__global__ __launch_bounds__(256)
__attribute__((amdgpu_waves_per_eu(2, 2)))
void gru_step_kernel(
    const __bf16* __restrict__ x,      // [64,1024] bf16
    const __bf16* __restrict__ hprev,  // [64,1024] bf16
    float* __restrict__ hf,            // [64,1024] f32 (read old / write new, col-owned)
    __bf16* __restrict__ hnext,        // [64,1024] bf16 out
    const __bf16* __restrict__ Wih,    // [3072,1024] bf16
    const __bf16* __restrict__ Whh,    // [3072,1024] bf16
    const float* __restrict__ bih, const float* __restrict__ bhh) {
  __shared__ float acc_s[4][B_][64];  // planes: s_r, s_z, gi_n, gh_n  (64 KB)
  const int tid  = threadIdx.x;
  const int lane = tid & 31;
  const int wv   = tid >> 5;        // 0..7
  const int j0   = blockIdx.x * 64; // hidden column base
  const int mt   = wv >> 1;         // 0..3
  const int nt0  = (wv & 1) * 2;    // 0 or 2
  const int lo16 = lane & 15;
  const int hi   = lane >> 4;
  const int m0   = mt * 16;
  const int nb   = j0 + nt0 * 16;   // column base within one gate block

  auto store_plane = [&](int plane, const v8f& a0, const v8f& a1) {
#pragma unroll
    for (int r = 0; r < 8; ++r) {
      int m = m0 + hi * 8 + r;
      acc_s[plane][m][nt0 * 16 + lo16]       = a0[r];
      acc_s[plane][m][(nt0 + 1) * 16 + lo16] = a1[r];
    }
  };

  v8f a0, a1;
  // gate r: s_r = x@Wih_r^T + h@Whh_r^T   (chained accumulation)
  a0 = zero8(); a1 = zero8();
  gemm_pass(x,     E_, Wih, E_, 0 * H_ + nb, m0, lane, a0, a1);
  gemm_pass(hprev, H_, Whh, H_, 0 * H_ + nb, m0, lane, a0, a1);
  store_plane(0, a0, a1);
  // gate z
  a0 = zero8(); a1 = zero8();
  gemm_pass(x,     E_, Wih, E_, 1 * H_ + nb, m0, lane, a0, a1);
  gemm_pass(hprev, H_, Whh, H_, 1 * H_ + nb, m0, lane, a0, a1);
  store_plane(1, a0, a1);
  // gate n: gi_n and gh_n kept separate (needed for i_n + r*h_n)
  a0 = zero8(); a1 = zero8();
  gemm_pass(x,     E_, Wih, E_, 2 * H_ + nb, m0, lane, a0, a1);
  store_plane(2, a0, a1);
  a0 = zero8(); a1 = zero8();
  gemm_pass(hprev, H_, Whh, H_, 2 * H_ + nb, m0, lane, a0, a1);
  store_plane(3, a0, a1);

  __syncthreads();

  // elementwise gate math on our 64x64 slice
  for (int i = tid; i < B_ * 64; i += 256) {
    int m = i >> 6, j = i & 63, jg = j0 + j;
    float sr  = acc_s[0][m][j] + bih[jg] + bhh[jg];
    float sz  = acc_s[1][m][j] + bih[H_ + jg] + bhh[H_ + jg];
    float gin = acc_s[2][m][j] + bih[2 * H_ + jg];
    float ghn = acc_s[3][m][j] + bhh[2 * H_ + jg];
    float rr = 1.0f / (1.0f + __expf(-sr));
    float zz = 1.0f / (1.0f + __expf(-sz));
    float nn = tanhf(gin + rr * ghn);
    float hold = hf[m * H_ + jg];
    float hn = (1.0f - zz) * nn + zz * hold;
    hf[m * H_ + jg]    = hn;
    hnext[m * H_ + jg] = (__bf16)hn;
  }
}

// ---------------------------------------------------------------------------
// logits tile [64 x 256] = h @ Wout^T slice, then per-row block-local argmax.
// h (128 KB) is async-DMA'd once into LDS (GLOBAL_LOAD_ASYNC_TO_LDS_B128,
// ASYNCcnt): 8 waves x 4 M-tiles reuse it, so the only global stream in the
// K-loop is W_out (the L2-resident 64 MB weight).
// Static LDS: 128K (h) + 64K (tile) + 2K (partials) < 320 KB per workgroup.
// ---------------------------------------------------------------------------
__global__ __launch_bounds__(256)
__attribute__((amdgpu_waves_per_eu(2, 2)))
void logits_argmax_kernel(
    const __bf16* __restrict__ h,      // [64,1024] bf16
    const __bf16* __restrict__ Wout,   // [32000,1024] bf16
    const float* __restrict__ bout,    // [32000]
    float* __restrict__ pmax,          // [NBLK_V][64]
    int* __restrict__ pidx) {
  __shared__ __bf16 sh[B_ * H_];     // 128 KB staged activations
  __shared__ float  lt[B_][NV_BLK];  // 64 KB logit tile
  __shared__ float  pr[4][B_];       // argmax partials
  __shared__ int    pri[4][B_];
  const int tid  = threadIdx.x;
  const int lane = tid & 31;
  const int wv   = tid >> 5;       // 0..7
  const int v0   = blockIdx.x * NV_BLK;
  const int n0   = wv * 32;        // this wave's two 16-wide n-tiles
  const int lo16 = lane & 15;
  const int hi   = lane >> 4;

  // async DMA stage of h into LDS: 16B per lane per issue, no VGPR round-trip
  {
    unsigned lds_base = (unsigned)(size_t)(&sh[0]);  // LDS byte offset of sh
    const uint4* s4 = (const uint4*)h;
#pragma unroll 1
    for (int i = tid; i < (B_ * H_) / 8; i += 256) {
      async_b128_to_lds(lds_base + (unsigned)i * 16u, s4 + i);
    }
    wait_asynccnt0();   // drain this wave's ASYNCcnt
  }
  __syncthreads();      // all waves' DMAs have landed -> LDS readable

  v8f acc[4][2];
#pragma unroll
  for (int m = 0; m < 4; ++m) { acc[m][0] = zero8(); acc[m][1] = zero8(); }

#pragma unroll 1
  for (int k0 = 0; k0 < H_; k0 += 32) {
    // prefetch next K-slab of this wave's weight columns
    __builtin_prefetch(Wout + (size_t)(v0 + n0 + lo16) * H_ + k0 + 32, 0, 0);
    v16bf b0 = load_b32x16(Wout, H_, v0 + n0,      k0, lane);
    v16bf b1 = load_b32x16(Wout, H_, v0 + n0 + 16, k0, lane);
#pragma unroll
    for (int m = 0; m < 4; ++m) {
      v16bf a = load_a16x32(sh, H_, m * 16, k0, lane);  // ds_load
      acc[m][0] = wmma_bf16(a, b0, acc[m][0]);
      acc[m][1] = wmma_bf16(a, b1, acc[m][1]);
    }
  }
#pragma unroll
  for (int m = 0; m < 4; ++m) {
#pragma unroll
    for (int i = 0; i < 2; ++i) {
#pragma unroll
      for (int r = 0; r < 8; ++r) {
        lt[m * 16 + hi * 8 + r][n0 + i * 16 + lo16] = acc[m][i][r];
      }
    }
  }
  __syncthreads();

  // parallel per-row argmax: 4 threads/row, each scans a 64-col quarter
  // (ascending order within and across quarters -> first-occurrence argmax)
  {
    int row = tid & 63, q = tid >> 6;
    float best = -3.0e38f; int bi = v0 + q * 64;
#pragma unroll 1
    for (int c = q * 64; c < q * 64 + 64; ++c) {
      float v = lt[row][c] + bout[v0 + c];
      if (v > best) { best = v; bi = v0 + c; }
    }
    pr[q][row] = best; pri[q][row] = bi;
  }
  __syncthreads();
  if (tid < B_) {
    float best = pr[0][tid]; int bi = pri[0][tid];
#pragma unroll
    for (int q = 1; q < 4; ++q) {
      if (pr[q][tid] > best) { best = pr[q][tid]; bi = pri[q][tid]; }
    }
    pmax[blockIdx.x * B_ + tid] = best;
    pidx[blockIdx.x * B_ + tid] = bi;
  }
}

// ---------------------------------------------------------------------------
// reduce block-local argmaxes (ascending v-order -> first occurrence), emit
// int64 token, gather+convert embedding row as next step's bf16 input.
// ---------------------------------------------------------------------------
__global__ void argmax_gather_kernel(const float* __restrict__ pmax,
                                     const int* __restrict__ pidx,
                                     const float* __restrict__ emb,  // fp32 [V,E]
                                     __bf16* __restrict__ x,         // [64,1024] bf16
                                     long long* __restrict__ resps, int t) {
  __shared__ int stok[B_];
  int tid = threadIdx.x;
  if (tid < B_) {
    float best = -3.0e38f; int bi = 0;
    for (int blk = 0; blk < NBLK_V; ++blk) {
      float v = pmax[blk * B_ + tid];
      if (v > best) { best = v; bi = pidx[blk * B_ + tid]; }
    }
    stok[tid] = bi;
    resps[(size_t)tid * T_ + t] = (long long)bi;
  }
  __syncthreads();
  for (int i = tid; i < B_ * E_; i += 256) {
    int b = i >> 10, e = i & (E_ - 1);
    x[i] = (__bf16)emb[(size_t)stok[b] * E_ + e];
  }
}

// resp_lens: first EOS index + 1, or T+1 if none (matches appended-ones argmax)
__global__ void resp_len_kernel(long long* __restrict__ out) {
  int b = threadIdx.x;
  if (b < B_) {
    long long len = T_ + 1;
    for (int t = 0; t < T_; ++t) {
      if (out[b * T_ + t] == (long long)EOS_TOK) { len = t + 1; break; }
    }
    out[B_ * T_ + b] = len;
  }
}

// ---------------------------------------------------------------------------
extern "C" void kernel_launch(void* const* d_in, const int* in_sizes, int n_in,
                              void* d_out, int out_size, void* d_ws, size_t ws_size,
                              hipStream_t stream) {
  (void)in_sizes; (void)n_in; (void)out_size; (void)ws_size;
  const float* zs   = (const float*)d_in[0];
  const float* cs   = (const float*)d_in[1];
  const float* emb  = (const float*)d_in[2];
  const float* Wih  = (const float*)d_in[3];
  const float* Whh  = (const float*)d_in[4];
  const float* bih  = (const float*)d_in[5];
  const float* bhh  = (const float*)d_in[6];
  const float* Wout = (const float*)d_in[7];
  const float* bout = (const float*)d_in[8];
  // d_in[9] = max_utt_len (device scalar) == 32 per reference setup; T_ hardcoded.

  char* ws = (char*)d_ws;
  size_t off = 0;
  auto take = [&](size_t bytes) -> void* {
    void* p = ws + off;
    off += (bytes + 255) & ~(size_t)255;
    return p;
  };
  __bf16* Wihb  = (__bf16*)take((size_t)H3_ * E_ * 2);
  __bf16* Whhb  = (__bf16*)take((size_t)H3_ * E_ * 2);
  __bf16* Woutb = (__bf16*)take((size_t)V_  * H_ * 2);
  __bf16* xb    = (__bf16*)take((size_t)B_  * E_ * 2);
  __bf16* hb0   = (__bf16*)take((size_t)B_  * H_ * 2);
  __bf16* hb1   = (__bf16*)take((size_t)B_  * H_ * 2);
  float*  hf    = (float*) take((size_t)B_  * H_ * 4);
  float*  pmax  = (float*) take((size_t)NBLK_V * B_ * 4);
  int*    pidx  = (int*)   take((size_t)NBLK_V * B_ * 4);

  auto cvt = [&](const float* s, __bf16* d, size_t n) {
    long n4 = (long)(n / 4);
    int grid = (int)((n4 + 255) / 256);
    cvt_f32_bf16_kernel<<<grid, 256, 0, stream>>>(s, d, n4);
  };
  cvt(Wih,  Wihb,  (size_t)H3_ * E_);
  cvt(Whh,  Whhb,  (size_t)H3_ * E_);
  cvt(Wout, Woutb, (size_t)V_  * H_);

  init_state_kernel<<<(B_ * H_) / 256, 256, 0, stream>>>(zs, cs, emb, hf, hb0, xb);

  long long* out = (long long*)d_out;
  __bf16* hb[2] = {hb0, hb1};
  for (int t = 0; t < T_; ++t) {
    const __bf16* hp = hb[t & 1];
    __bf16*       hn = hb[1 - (t & 1)];
    gru_step_kernel<<<H_ / 64, 256, 0, stream>>>(xb, hp, hf, hn, Wihb, Whhb, bih, bhh);
    logits_argmax_kernel<<<NBLK_V, 256, 0, stream>>>(hn, Woutb, bout, pmax, pidx);
    argmax_gather_kernel<<<1, 256, 0, stream>>>(pmax, pidx, emb, xb, out, t);
  }
  resp_len_kernel<<<1, 64, 0, stream>>>(out);
}
